// HumanReceiver_69801808494805
// MI455X (gfx1250) — matrix operand order repaired and checked by reading, not compiled, not run on hardware
//
#include <hip/hip_runtime.h>
#include <hip/hip_bf16.h>

typedef __attribute__((ext_vector_type(16))) _Float16 v16h;
typedef __attribute__((ext_vector_type(8)))  _Float16 v8h;
typedef __attribute__((ext_vector_type(8)))  float    v8f;
typedef __attribute__((ext_vector_type(4)))  int      v4i;
typedef __attribute__((address_space(1))) v4i gv4i;   // global int4
typedef __attribute__((address_space(3))) v4i lv4i;   // LDS int4

#define GQ   512
#define PQ   64
#define EPGQ 1024
#define NQ   (GQ * PQ)        // 32768
#define EQ   (GQ * EPGQ)      // 524288
#define LQ   10
#define EMBQ 128
#define HQ   256
#define H3Q  768
#define RQ   4
#define FKQ  8

#if defined(__gfx1250__) && __has_builtin(__builtin_amdgcn_global_load_async_to_lds_b128) && __has_builtin(__builtin_amdgcn_s_wait_asynccnt)
#define USE_ASYNC_LDS 1
#else
#define USE_ASYNC_LDS 0
#endif

// D = A(16x32 f16) * B(32x16 f16) + C(16x16 f32)   [CDNA5 wave32 WMMA]
__device__ __forceinline__ v8f wmma16(v16h a, v16h b, v8f c) {
  return __builtin_amdgcn_wmma_f32_16x16x32_f16(false, a, false, b,
                                                (short)0, c, false, false);
}

// Build a 16x32 f16 A fragment from a row-major f16 LDS row:
// halves 0..7 = K[kb..kb+7], halves 8..15 = K[kb+16..kb+23], kb = kc*32+(hi?8:0)
__device__ __forceinline__ v16h a_frag(const _Float16* row, int kc, bool hiL) {
  const v8h* ap = (const v8h*)row;
  int base = (kc * 32 + (hiL ? 8 : 0)) >> 3;   // in v8h units
  v8h lo = ap[base], hi = ap[base + 2];
  return __builtin_shufflevector(lo, hi, 0, 1, 2, 3, 4, 5, 6, 7,
                                 8, 9, 10, 11, 12, 13, 14, 15);
}

// ---------------------------------------------------------------------------
// Kernel 0: pack W_ih / W_hh / fc_W into f16 B-fragment-native layout:
//   P[((nt*nkc + kc)*32 + lane)*16 + j] = W[(kc*32 + ((j&8)<<1) + ((lane&16)>>1) + (j&7))*Nc + nt*16 + (lane&15)]
// so each lane's B fragment is one contiguous 32B v16h load.
// ---------------------------------------------------------------------------
__device__ __forceinline__ void pack_frag(int idx, const float* __restrict__ W,
                                          _Float16* __restrict__ P,
                                          int Nc, int kcBits) {
  int j    = idx & 15;
  int lane = (idx >> 4) & 31;
  int frag = idx >> 9;
  int kc   = frag & ((1 << kcBits) - 1);
  int nt   = frag >> kcBits;
  int k = (kc << 5) + ((j & 8) << 1) + ((lane & 16) >> 1) + (j & 7);
  int n = (nt << 4) + (lane & 15);
  P[idx] = (_Float16)W[k * Nc + n];
}

__global__ void __launch_bounds__(256) pack_kernel(
    const float* __restrict__ Wih, const float* __restrict__ Whh,
    const float* __restrict__ fcW, _Float16* __restrict__ pWih,
    _Float16* __restrict__ pWhh, _Float16* __restrict__ pfcW)
{
  const int SZ_IH = EMBQ * H3Q;   // 98304
  const int SZ_HH = HQ * H3Q;     // 196608
  const int SZ_FC = HQ * EMBQ;    // 32768
  int idx = blockIdx.x * 256 + threadIdx.x;
  if (idx < SZ_IH)                    pack_frag(idx, Wih, pWih, H3Q, 2);
  else if (idx < SZ_IH + SZ_HH)       pack_frag(idx - SZ_IH, Whh, pWhh, H3Q, 3);
  else if (idx < SZ_IH + SZ_HH + SZ_FC) pack_frag(idx - SZ_IH - SZ_HH, fcW, pfcW, EMBQ, 3);
}

// ---------------------------------------------------------------------------
// Kernel 1: per-group RGCN, fully in LDS (320KB/WGP on MI455X).
// grid = 512 (one group per block), 256 threads = 8 wave32 waves.
// W_rel packed into B-fragment layout in LDS (K=8 zero-padded to 32 baked
// in); xs A-fragments prebuilt in registers with nt as a compile-time index,
// so the tile loop is: one 32B ds load -> v_wmma -> D store.
// ~200 KB dynamic LDS; the E*128-float edge scatter never leaves the WGP.
// ---------------------------------------------------------------------------
__global__ void __launch_bounds__(256) rgcn_kernel(
    const float* __restrict__ x, const int* __restrict__ ei,
    const int* __restrict__ etype, const float* __restrict__ Wrel,
    const float* __restrict__ Wroot, const float* __restrict__ brg,
    float* __restrict__ hout)
{
  extern __shared__ float smf[];
  float*     s_xw    = smf;                  // 64*512 f32
  float*     s_agg   = s_xw + 64 * 512;      // 64*128 f32
  float*     s_xs    = s_agg + 64 * 128;     // 64*8   f32
  float*     s_wroot = s_xs + 64 * 8;        // 8*128  f32
  float*     s_b     = s_wroot + 8 * 128;    // 128    f32
  float*     s_cnt   = s_b + 128;            // 256    f32
  _Float16*  s_xsh   = (_Float16*)(s_cnt + 256);   // 64*8 f16
  _Float16*  s_wrelp = s_xsh + 64 * FKQ;           // 32 frags * 32 * 16 f16

  const int tid = threadIdx.x;
  const int g   = blockIdx.x;

  for (int i = tid; i < FKQ * EMBQ; i += 256) s_wroot[i] = Wroot[i];
  for (int i = tid; i < EMBQ; i += 256)       s_b[i] = brg[i];
  for (int i = tid; i < 64 * FKQ; i += 256) {
    int n = i >> 3, f = i & 7;
    float v = x[(g * 64 + n) * 16 + f];              // xs = x[:, :8]
    s_xs[i]  = v;
    s_xsh[i] = (_Float16)v;
  }
  // W_rel -> B-fragment layout, frag = r*8+ct, zero-padded K
  for (int i = tid; i < RQ * 8 * 32 * 16; i += 256) {
    int j = i & 15, ln = (i >> 4) & 31, frag = i >> 9;
    int r = frag >> 3, ct = frag & 7;
    _Float16 v = (_Float16)0.0f;
    if (ln < 16 && j < FKQ)
      v = (_Float16)Wrel[r * (FKQ * EMBQ) + j * EMBQ + ct * 16 + ln];
    s_wrelp[i] = v;
  }
  for (int i = tid; i < 64 * EMBQ; i += 256) s_agg[i] = 0.0f;
  for (int i = tid; i < 64 * RQ; i += 256)   s_cnt[i] = 0.0f;
  __syncthreads();

  const int ebase = g * EPGQ;

  // degree counts per (dst, rel) -- LDS float atomics
  for (int e = tid; e < EPGQ; e += 256) {
    int d = ei[EQ + ebase + e] - g * 64;
    int r = etype[ebase + e];
    atomicAdd(&s_cnt[d * RQ + r], 1.0f);
  }

  const int wave = tid >> 5, lane = tid & 31;
  const int m16  = lane & 15;
  const bool hiL = (lane >= 16);

  // Prebuild the 4 xs A-fragments (one per 16-node tile), zero-padded K
  v8h zed = {};
  v16h axs[4];
#pragma unroll
  for (int nt = 0; nt < 4; ++nt) {
    v8h lo = zed;
    if (!hiL) lo = *(const v8h*)(s_xsh + (nt * 16 + m16) * FKQ);
    axs[nt] = __builtin_shufflevector(lo, zed, 0, 1, 2, 3, 4, 5, 6, 7,
                                      8, 9, 10, 11, 12, 13, 14, 15);
  }

  // xW[node, rel, :] = xs(64x8) @ W_rel[r](8x128): 128 tiles.
  // nt is the outer *unrolled* loop so axs[nt] is a compile-time index.
  const v16h* brel = (const v16h*)s_wrelp;
  const int row0 = hiL ? 8 : 0;
#pragma unroll
  for (int nt = 0; nt < 4; ++nt) {
    for (int u = wave; u < 32; u += 8) {        // u = r*8 + ct
      int r = u >> 3, ct = u & 7;
      v8f c = {};
      c = wmma16(axs[nt], brel[u * 32 + lane], c);
#pragma unroll
      for (int v = 0; v < 8; ++v)
        s_xw[(nt * 16 + row0 + v) * (RQ * EMBQ) + r * EMBQ + ct * 16 + m16] = c[v];
    }
  }
  __syncthreads();

  // normalized scatter: agg[dst] += xW[src, et] / cnt   (all in LDS)
  for (int e = wave; e < EPGQ; e += 8) {
    int s = ei[ebase + e] - g * 64;
    int d = ei[EQ + ebase + e] - g * 64;
    int r = etype[ebase + e];
    float nrm = __builtin_amdgcn_rcpf(fmaxf(s_cnt[d * RQ + r], 1.0f));
    const float* srcrow = s_xw + s * (RQ * EMBQ) + r * EMBQ;
    for (int cc = lane; cc < EMBQ; cc += 32)
      atomicAdd(&s_agg[d * EMBQ + cc], srcrow[cc] * nrm);
  }
  __syncthreads();

  // h = relu(xs @ W_root + agg + b)
  for (int idx = tid; idx < 64 * EMBQ; idx += 256) {
    int n = idx >> 7, cc = idx & 127;
    float acc = s_b[cc] + s_agg[idx];
#pragma unroll
    for (int f = 0; f < FKQ; ++f) acc += s_xs[n * FKQ + f] * s_wroot[f * EMBQ + cc];
    hout[(g * 64 + n) * EMBQ + cc] = fmaxf(acc, 0.0f);
  }
}

// ---------------------------------------------------------------------------
// Kernel 2: GI = embed[message] @ W_ih + b_ih   (5120 x 128 x 768 WMMA GEMM)
// grid = 320 blocks (one 16-row M-tile), 8 waves x 6 N-tiles.
// Embedding rows gathered to LDS via GLOBAL_LOAD_ASYNC_TO_LDS_B128
// (ASYNCcnt path); B fragments are single 32B loads from pWih.
// ---------------------------------------------------------------------------
__global__ void __launch_bounds__(256) gi_kernel(
    const int* __restrict__ msg, const float* __restrict__ etab,
    const _Float16* __restrict__ pWih, const float* __restrict__ bih,
    float* __restrict__ giall)
{
  __shared__ _Float16 s_ah[16 * EMBQ];
  const int tid = threadIdx.x;
  const int mt  = blockIdx.x;

#if USE_ASYNC_LDS
  __shared__ float s_af[16 * EMBQ];
  {
    int rowi = tid >> 4;                 // 0..15
    int k    = (tid & 15) * 8;           // 0..120
    const float* src = etab + msg[mt * 16 + rowi] * EMBQ + k;
    float*       dst = s_af + rowi * EMBQ + k;
    __builtin_amdgcn_global_load_async_to_lds_b128(
        (gv4i*)src, (lv4i*)dst, 0, 0);
    __builtin_amdgcn_global_load_async_to_lds_b128(
        (gv4i*)(src + 4), (lv4i*)(dst + 4), 0, 0);
    __builtin_amdgcn_s_wait_asynccnt(0);
  }
  __syncthreads();
  for (int i = tid; i < 16 * EMBQ; i += 256) s_ah[i] = (_Float16)s_af[i];
#else
  for (int i = tid; i < 16 * EMBQ; i += 256) {
    int mrow = i >> 7, k = i & 127;
    s_ah[i] = (_Float16)etab[msg[mt * 16 + mrow] * EMBQ + k];
  }
#endif
  __syncthreads();

  const int wave = tid >> 5, lane = tid & 31;
  const int m16  = lane & 15;
  const bool hiL = (lane >= 16);
  const v16h* bp = (const v16h*)pWih;
  const _Float16* arow = s_ah + m16 * EMBQ;

  v16h afrag[4];
#pragma unroll
  for (int kc = 0; kc < 4; ++kc) afrag[kc] = a_frag(arow, kc, hiL);

  for (int i = 0; i < 6; ++i) {
    int nt = wave * 6 + i, n0 = nt * 16;
    v8f acc = {};
#pragma unroll
    for (int kc = 0; kc < 4; ++kc)
      acc = wmma16(afrag[kc], bp[(nt * 4 + kc) * 32 + lane], acc);
    int row0 = hiL ? 8 : 0;
#pragma unroll
    for (int v = 0; v < 8; ++v)
      giall[(mt * 16 + row0 + v) * H3Q + n0 + m16] = acc[v] + bih[n0 + m16];
  }
}

// ---------------------------------------------------------------------------
// Kernel 3: GRU recurrence. grid = 32 blocks x 16 groups, 8 waves.
// Hidden kept in LDS twice: f32 (exact update) + f16 mirror (A fragments as
// two ds_load_b128).  W_hh fragments: one 32B load each, L2-resident.
// ---------------------------------------------------------------------------
__global__ void __launch_bounds__(256) gru_kernel(
    const float* __restrict__ giall, const _Float16* __restrict__ pWhh,
    const float* __restrict__ bhh, const _Float16* __restrict__ pfcW,
    const float* __restrict__ fcb, float* __restrict__ hproj)
{
  extern __shared__ float smg[];
  float*    s_h    = smg;                         // 16*256 f32
  float*    s_gh   = smg + 16 * HQ;               // 16*768 f32
  _Float16* s_hf16 = (_Float16*)(s_gh + 16 * H3Q); // 16*256 f16

  const int tid = threadIdx.x, wave = tid >> 5, lane = tid & 31;
  const int m16 = lane & 15;
  const bool hiL = (lane >= 16);
  const int gbase = blockIdx.x * 16;
  const v16h* bph = (const v16h*)pWhh;
  const v16h* bpf = (const v16h*)pfcW;
  const _Float16* arow = s_hf16 + m16 * HQ;

  for (int i = tid; i < 16 * HQ; i += 256) { s_h[i] = 0.0f; s_hf16[i] = (_Float16)0.0f; }
  __syncthreads();

  for (int t = 0; t < LQ; ++t) {
    // gh = h @ W_hh + b_hh  (48 N-tiles x 8 K-chunks, 6 N-tiles per wave)
    for (int i = 0; i < 6; ++i) {
      int nt = wave * 6 + i, n0 = nt * 16;
      if (nt + 1 < 48)
        __builtin_prefetch((const void*)(bph + (nt + 1) * 8 * 32 + lane), 0, 0);
      v8f acc = {};
#pragma unroll
      for (int kc = 0; kc < 8; ++kc)
        acc = wmma16(a_frag(arow, kc, hiL), bph[(nt * 8 + kc) * 32 + lane], acc);
      int row0 = hiL ? 8 : 0;
#pragma unroll
      for (int v = 0; v < 8; ++v)
        s_gh[(row0 + v) * H3Q + n0 + m16] = acc[v] + bhh[n0 + m16];
    }
    __syncthreads();

    // gates: r = sig(ir+hr), z = sig(iz+hz), n = tanh(in + r*hn)
    for (int i = tid; i < 16 * HQ; i += 256) {
      int m = i >> 8, j = i & 255;
      const float* gi = giall + ((gbase + m) * LQ + t) * H3Q;
      float ghr = s_gh[m * H3Q + j];
      float ghz = s_gh[m * H3Q + 256 + j];
      float ghn = s_gh[m * H3Q + 512 + j];
      float r = 1.0f / (1.0f + __expf(-(gi[j] + ghr)));
      float z = 1.0f / (1.0f + __expf(-(gi[256 + j] + ghz)));
      float n = tanhf(gi[512 + j] + r * ghn);
      float hn = (1.0f - z) * n + z * s_h[i];
      s_h[i] = hn;
      s_hf16[i] = (_Float16)hn;
    }
    __syncthreads();
  }

  // hproj = h @ fc_W + fc_b : 16x256 @ 256x128, one N-tile per wave
  {
    int n0 = wave * 16;
    v8f acc = {};
#pragma unroll
    for (int kc = 0; kc < 8; ++kc)
      acc = wmma16(a_frag(arow, kc, hiL), bpf[(wave * 8 + kc) * 32 + lane], acc);
    int row0 = hiL ? 8 : 0;
#pragma unroll
    for (int v = 0; v < 8; ++v)
      hproj[(gbase + row0 + v) * EMBQ + n0 + m16] = acc[v] + fcb[n0 + m16];
  }
}

// ---------------------------------------------------------------------------
// Kernel 4: per-group logits, log-softmax, entropy, Gumbel argmax sample.
// ---------------------------------------------------------------------------
__global__ void __launch_bounds__(64) sample_kernel(
    const float* __restrict__ hbuf, const float* __restrict__ hproj,
    float* __restrict__ out)
{
  __shared__ float s_hp[EMBQ];
  __shared__ float s_logit[PQ];
  __shared__ float s_red[PQ];
  __shared__ float s_val[PQ];
  __shared__ int   s_idx[PQ];

  const int g = blockIdx.x, p = threadIdx.x;
  for (int i = p; i < EMBQ; i += PQ) s_hp[i] = hproj[g * EMBQ + i];
  __syncthreads();

  float acc = 0.0f;
  const float* hr = hbuf + (g * PQ + p) * EMBQ;
  for (int c = 0; c < EMBQ; ++c) acc += hr[c] * s_hp[c];
  s_logit[p] = acc;

  // deterministic hash-based Gumbel (JAX threefry not reproducible here)
  unsigned u = (unsigned)(g * PQ + p);
  u ^= 2747636419u; u *= 2654435769u; u ^= u >> 16;
  u *= 2654435769u; u ^= u >> 16; u *= 2654435769u;
  float uf  = (float)(u >> 8) * (1.0f / 16777216.0f) + 1e-12f;
  float gum = -__logf(-__logf(uf));

  s_red[p] = acc;
  __syncthreads();
  for (int s = 32; s > 0; s >>= 1) {
    if (p < s) s_red[p] = fmaxf(s_red[p], s_red[p + s]);
    __syncthreads();
  }
  float mx = s_red[0];
  __syncthreads();
  s_red[p] = __expf(acc - mx);
  __syncthreads();
  for (int s = 32; s > 0; s >>= 1) {
    if (p < s) s_red[p] += s_red[p + s];
    __syncthreads();
  }
  float lse = mx + __logf(s_red[0]);
  __syncthreads();
  float logp = acc - lse;
  s_red[p] = -__expf(logp) * logp;
  __syncthreads();
  for (int s = 32; s > 0; s >>= 1) {
    if (p < s) s_red[p] += s_red[p + s];
    __syncthreads();
  }
  float ent = s_red[0];
  s_val[p] = acc + gum;
  s_idx[p] = p;
  __syncthreads();
  for (int s = 32; s > 0; s >>= 1) {
    if (p < s && s_val[p + s] > s_val[p]) {
      s_val[p] = s_val[p + s];
      s_idx[p] = s_idx[p + s];
    }
    __syncthreads();
  }
  if (p == 0) {
    int samp = s_idx[0];
    out[g]          = (float)samp;            // sample
    out[GQ + g]     = s_logit[samp] - lse;    // log_prob
    out[2 * GQ + g] = ent;                    // entropy
  }
}

// ---------------------------------------------------------------------------
extern "C" void kernel_launch(void* const* d_in, const int* in_sizes, int n_in,
                              void* d_out, int out_size, void* d_ws, size_t ws_size,
                              hipStream_t stream)
{
  (void)in_sizes; (void)n_in; (void)out_size; (void)ws_size;
  const int*   msg   = (const int*)  d_in[0];
  const float* x     = (const float*)d_in[1];
  const int*   ei    = (const int*)  d_in[2];
  const int*   et    = (const int*)  d_in[3];
  const float* Wrel  = (const float*)d_in[4];
  const float* Wroot = (const float*)d_in[5];
  const float* brg   = (const float*)d_in[6];
  const float* etab  = (const float*)d_in[7];
  const float* Wih   = (const float*)d_in[8];
  const float* Whh   = (const float*)d_in[9];
  const float* bih   = (const float*)d_in[10];
  const float* bhh   = (const float*)d_in[11];
  const float* fcW   = (const float*)d_in[12];
  const float* fcb   = (const float*)d_in[13];
  float* out = (float*)d_out;

  // workspace (~33 MB): hbuf | giall | hproj | packed f16 weights
  char*  ws    = (char*)d_ws;
  size_t off   = 0;
  float* hbuf  = (float*)(ws + off);  off += (size_t)NQ * EMBQ * 4;
  float* giall = (float*)(ws + off);  off += (size_t)(GQ * LQ) * H3Q * 4;
  float* hproj = (float*)(ws + off);  off += (size_t)GQ * EMBQ * 4;
  _Float16* pWih = (_Float16*)(ws + off); off += (size_t)EMBQ * H3Q * 2;
  _Float16* pWhh = (_Float16*)(ws + off); off += (size_t)HQ * H3Q * 2;
  _Float16* pfcW = (_Float16*)(ws + off); off += (size_t)HQ * EMBQ * 2;

  const size_t lds_rgcn =
      (size_t)(64 * 512 + 64 * 128 + 64 * 8 + 8 * 128 + 128 + 256) * 4 +
      (size_t)(64 * FKQ + RQ * 8 * 32 * 16) * 2;          // ~200 KB (<320 KB)
  const size_t lds_gru =
      (size_t)(16 * HQ + 16 * H3Q) * 4 + (size_t)(16 * HQ) * 2;  // 72 KB

  const int packN = (EMBQ * H3Q + HQ * H3Q + HQ * EMBQ + 255) / 256;

  rgcn_kernel<<<GQ, 256, lds_rgcn, stream>>>(x, ei, et, Wrel, Wroot, brg, hbuf);
  pack_kernel<<<packN, 256, 0, stream>>>(Wih, Whh, fcW, pWih, pWhh, pfcW);
  gi_kernel<<<(GQ * LQ) / 16, 256, 0, stream>>>(msg, etab, pWih, bih, giall);
  gru_kernel<<<GQ / 16, 256, lds_gru, stream>>>(giall, pWhh, bhh, pfcW, fcb, hproj);
  sample_kernel<<<GQ, PQ, 0, stream>>>(hbuf, hproj, out);
}
